// DifferentialMultiHeadCrossAttention_73564199846398
// MI455X (gfx1250) — matrix operand order repaired and checked by reading, not compiled, not run on hardware
//
#include <hip/hip_runtime.h>

// ---------------------------------------------------------------------------
// CDNA5 (gfx1250) differential multi-head cross-attention, bf16 WMMA pipeline.
//   - all GEMMs on v_wmma_f32_16x16x32_bf16 (8x the MACs/instr of fp32 WMMA)
//   - flash-style attention, two concurrent base-2 online softmaxes
//   - softmax scale (0.125*log2e) folded into the q projections
//   - LDS tile staging via GLOBAL_LOAD_ASYNC_TO_LDS_B128 (ASYNCcnt DMA path)
// ---------------------------------------------------------------------------

typedef __attribute__((ext_vector_type(16))) __bf16 v16bf;
typedef __attribute__((ext_vector_type(8)))  float  v8f;

namespace cfg {
constexpr int Bsz = 4, TQ = 1024, TK = 1024, E = 1024, H = 16, DH = 64;
constexpr int HD   = H * DH;        // 1024
constexpr int Mtok = Bsz * TQ;      // 4096
}

__device__ inline __bf16 f2bf(float f) {
  unsigned u = __builtin_bit_cast(unsigned, f);
  u += 0x7FFFu + ((u >> 16) & 1u);            // round-to-nearest-even
  unsigned short hv = (unsigned short)(u >> 16);
  return __builtin_bit_cast(__bf16, hv);
}

__device__ inline v8f vzero8() {
  v8f z;
#pragma unroll
  for (int i = 0; i < 8; ++i) z[i] = 0.f;
  return z;
}

// Async DMA: 16 bytes global -> LDS, tracked by ASYNCcnt.
// LDS byte address = low 32 bits of the generic pointer (LDS aperture keeps
// the offset in addr[31:0]); global address as 64-bit VGPR pair, GV mode.
__device__ inline void async_ld_b128(unsigned lds, const void* g) {
  asm volatile("global_load_async_to_lds_b128 %0, %1, off"
               :: "v"(lds), "v"((unsigned long long)(size_t)g)
               : "memory");
}
__device__ inline void wait_async0() {
  asm volatile("s_wait_asynccnt 0x0" ::: "memory");
}

// A-matrix 16x32 bf16 fragment (row-major source row, any addr space).
// element j: K = hf*8 + j (j<8) ; K = 16 + hf*8 + (j-8) (j>=8)
__device__ inline v16bf ld_fragA(const __bf16* row, int hf) {
  v16bf f;
  const __bf16* p0 = row + hf * 8;
  const __bf16* p1 = row + 16 + hf * 8;
#pragma unroll
  for (int j = 0; j < 8; ++j) { f[j] = p0[j]; f[8 + j] = p1[j]; }
  return f;
}

// B-matrix 32x16 bf16 fragment (N-major source row = one column of B).
// element j: K = hf*8 + j (j<8) ; K = 16 + hf*8 + (j-8) (j>=8)
__device__ inline v16bf ld_fragB(const __bf16* row, int hf) {
  v16bf f;
  const __bf16* p0 = row + hf * 8;
  const __bf16* p1 = row + 16 + hf * 8;
#pragma unroll
  for (int j = 0; j < 8; ++j) { f[j] = p0[j]; f[8 + j] = p1[j]; }
  return f;
}

__device__ inline v8f wmma_bf16(v16bf a, v16bf b, v8f c) {
  return __builtin_amdgcn_wmma_f32_16x16x32_bf16(false, a, false, b,
                                                 (short)0, c, false, false);
}

// ---------------------------------------------------------------------------
// fp32 -> bf16 conversion
// ---------------------------------------------------------------------------
__global__ void __launch_bounds__(256)
cvt_f32_bf16(const float* __restrict__ src, __bf16* __restrict__ dst, int n) {
  int i = blockIdx.x * 256 + threadIdx.x;
  if (i < n) dst[i] = f2bf(src[i]);
}

// ---------------------------------------------------------------------------
// 128x128 tile GEMM:  C[m,n] = (sum_k A[m,k] * Bw[n,k] + bias[n]) * scale
// A: [4096 x 1024] bf16 row-major, Bw: [1024 x 1024] bf16 row-major (N x K)
// MODE 0: bf16 out in [B,H,T,DH] head layout (projections)
// MODE 1: fp32 out row-major [m, n]           (output projection)
// 8 waves; wave tile 32(M) x 64(N) = 2x4 WMMA accumulators.
// K tiles staged into LDS with async b128 DMA (4 per thread per K-step).
// ---------------------------------------------------------------------------
template <int MODE>
__global__ void __launch_bounds__(256)
gemm_bf16_128x128(const __bf16* __restrict__ A, const __bf16* __restrict__ Bw,
                  const float* __restrict__ bias, void* __restrict__ Cout,
                  float scale) {
  constexpr int K = 1024, N = 1024;
  __shared__ __bf16 Asm[128 * 32];
  __shared__ __bf16 Bsm[128 * 32];

  const int tid = threadIdx.x;
  const int lane = tid & 31, wid = tid >> 5;
  const int lane16 = lane & 15, hf = lane >> 4;
  const int waveM = wid & 3, waveN = wid >> 2;   // 4 x 2 wave grid
  const int m0 = blockIdx.y * 128, n0 = blockIdx.x * 128;

  v8f acc[2][4];
#pragma unroll
  for (int mi = 0; mi < 2; ++mi)
#pragma unroll
    for (int ni = 0; ni < 4; ++ni) acc[mi][ni] = vzero8();

  // cooperative loader: each thread DMAs 16 contiguous bf16 (2 x b128) / tile
  const int crow = tid >> 1, ccol = (tid & 1) * 16;
  const __bf16* aSrc = A + (size_t)(m0 + crow) * K + ccol;
  const __bf16* bSrc = Bw + (size_t)(n0 + crow) * K + ccol;
  const unsigned aDst = (unsigned)(size_t)&Asm[crow * 32 + ccol];
  const unsigned bDst = (unsigned)(size_t)&Bsm[crow * 32 + ccol];

  for (int k0 = 0; k0 < K; k0 += 32) {
    __syncthreads();
    async_ld_b128(aDst,      aSrc + k0);
    async_ld_b128(aDst + 16, aSrc + k0 + 8);
    async_ld_b128(bDst,      bSrc + k0);
    async_ld_b128(bDst + 16, bSrc + k0 + 8);
    if (k0 + 32 < K) {                       // pull next K-slab toward L2
      __builtin_prefetch(aSrc + k0 + 32, 0, 1);
      __builtin_prefetch(bSrc + k0 + 32, 0, 1);
    }
    wait_async0();
    __syncthreads();

    v16bf af[2], bfm[4];
#pragma unroll
    for (int mi = 0; mi < 2; ++mi)
      af[mi] = ld_fragA(&Asm[(waveM * 32 + mi * 16 + lane16) * 32], hf);
#pragma unroll
    for (int ni = 0; ni < 4; ++ni)
      bfm[ni] = ld_fragB(&Bsm[(waveN * 64 + ni * 16 + lane16) * 32], hf);
#pragma unroll
    for (int mi = 0; mi < 2; ++mi)
#pragma unroll
      for (int ni = 0; ni < 4; ++ni)
        acc[mi][ni] = wmma_bf16(af[mi], bfm[ni], acc[mi][ni]);
  }

#pragma unroll
  for (int mi = 0; mi < 2; ++mi) {
#pragma unroll
    for (int ni = 0; ni < 4; ++ni) {
      const int nn = n0 + waveN * 64 + ni * 16 + lane16;
      const float bv = bias[nn];
#pragma unroll
      for (int r = 0; r < 8; ++r) {
        const int mm = m0 + waveM * 32 + mi * 16 + r + 8 * hf;
        const float v = (acc[mi][ni][r] + bv) * scale;
        if constexpr (MODE == 0) {
          const int bb = mm >> 10, tt = mm & 1023;
          const int hh = nn >> 6,  dd = nn & 63;
          ((__bf16*)Cout)[(((size_t)(bb * cfg::H + hh)) * cfg::TQ + tt) * cfg::DH + dd] = f2bf(v);
        } else {
          ((float*)Cout)[(size_t)mm * N + nn] = v;
        }
      }
    }
  }
}

// ---------------------------------------------------------------------------
// base-2 online-softmax update for one 16x32 score tile pair (per wave).
// Scores arrive pre-scaled by 0.125*log2(e). Row M = r + 8*hf lives in the
// 16 lanes of group `hf`; xor masks 1..8 reduce within that group.
// Writes exp2(s - m_new) to per-wave LDS in [M][s] order.
// ---------------------------------------------------------------------------
__device__ inline void softmax_update(v8f st[2], float* m, float* l, v8f* o,
                                      __bf16* pLDS, int lane16, int hf) {
#pragma unroll
  for (int r = 0; r < 8; ++r) {
    float cmax = fmaxf(st[0][r], st[1][r]);
#pragma unroll
    for (int mk = 1; mk < 16; mk <<= 1) cmax = fmaxf(cmax, __shfl_xor(cmax, mk, 32));
    const float mnew = fmaxf(m[r], cmax);
    const float p0 = exp2f(st[0][r] - mnew);
    const float p1 = exp2f(st[1][r] - mnew);
    pLDS[(r + 8 * hf) * 32 + lane16]      = f2bf(p0);
    pLDS[(r + 8 * hf) * 32 + 16 + lane16] = f2bf(p1);
    float rs = p0 + p1;
#pragma unroll
    for (int mk = 1; mk < 16; mk <<= 1) rs += __shfl_xor(rs, mk, 32);
    const float alpha = exp2f(m[r] - mnew);
    l[r] = l[r] * alpha + rs;
    m[r] = mnew;
#pragma unroll
    for (int dt = 0; dt < 4; ++dt) o[dt][r] *= alpha;
  }
}

// ---------------------------------------------------------------------------
// flash-style differential attention: grid (B*H, TQ/128), 256 threads.
// Each wave owns 16 query rows; loop over TK in 32-key chunks.
// k1/k2 tiles staged by async DMA; v staged transposed via ds stores.
// ---------------------------------------------------------------------------
__global__ void __launch_bounds__(256)
diff_attn_kernel(const __bf16* __restrict__ q1, const __bf16* __restrict__ q2,
                 const __bf16* __restrict__ k1, const __bf16* __restrict__ k2,
                 const __bf16* __restrict__ v,  const float* __restrict__ lam,
                 float* __restrict__ out) {
  using namespace cfg;
  __shared__ __bf16 k1s[32 * 64];
  __shared__ __bf16 k2s[32 * 64];
  __shared__ __bf16 vTs[64 * 32];          // transposed: [d][s]
  __shared__ __bf16 pbuf[8][2][16 * 32];   // per-wave p1/p2 transpose scratch

  const int tid = threadIdx.x;
  const int lane = tid & 31, wid = tid >> 5;
  const int lane16 = lane & 15, hf = lane >> 4;
  const int bh = blockIdx.x;
  const float lam_h = lam[bh & (H - 1)];
  const size_t hb = (size_t)bh;
  const __bf16* q1h = q1 + hb * TQ * DH;
  const __bf16* q2h = q2 + hb * TQ * DH;
  const __bf16* k1h = k1 + hb * TK * DH;
  const __bf16* k2h = k2 + hb * TK * DH;
  const __bf16* vh  = v  + hb * TK * DH;
  float* outh = out + hb * TQ * DH;

  const int qrow0 = blockIdx.y * 128 + wid * 16;

  v16bf q1f[2], q2f[2];
#pragma unroll
  for (int kk = 0; kk < 2; ++kk) {
    q1f[kk] = ld_fragA(q1h + (size_t)(qrow0 + lane16) * DH + kk * 32, hf);
    q2f[kk] = ld_fragA(q2h + (size_t)(qrow0 + lane16) * DH + kk * 32, hf);
  }

  float m1[8], l1[8], m2[8], l2[8];
  v8f o1[4], o2[4];
#pragma unroll
  for (int r = 0; r < 8; ++r) { m1[r] = m2[r] = -1e30f; l1[r] = l2[r] = 0.f; }
#pragma unroll
  for (int dt = 0; dt < 4; ++dt) { o1[dt] = vzero8(); o2[dt] = vzero8(); }

  const int cs = tid >> 3;           // source key row 0..31
  const int cd = (tid & 7) << 3;     // d offset 0..56 (8 bf16 = 16 bytes)
  const unsigned k1Dst = (unsigned)(size_t)&k1s[cs * 64 + cd];
  const unsigned k2Dst = (unsigned)(size_t)&k2s[cs * 64 + cd];

  for (int kc = 0; kc < TK; kc += 32) {
    __syncthreads();                 // previous-iteration LDS reads done
    async_ld_b128(k1Dst, k1h + (size_t)(kc + cs) * DH + cd);
    async_ld_b128(k2Dst, k2h + (size_t)(kc + cs) * DH + cd);
    {
      const __bf16* svp = vh + (size_t)(kc + cs) * DH + cd;
#pragma unroll
      for (int j = 0; j < 8; ++j) vTs[(cd + j) * 32 + cs] = svp[j];
    }
    wait_async0();
    __syncthreads();

    // s = q . k^T  (K-dim = DH = 64 -> two WMMAs per 16-col tile);
    // scale already folded into q, so accumulators are used directly.
    v8f st1[2], st2[2];
#pragma unroll
    for (int t = 0; t < 2; ++t) {
      v8f a1 = vzero8(), a2 = vzero8();
      const __bf16* r1 = &k1s[(t * 16 + lane16) * 64];
      const __bf16* r2 = &k2s[(t * 16 + lane16) * 64];
      a1 = wmma_bf16(q1f[0], ld_fragB(r1, hf), a1);
      a1 = wmma_bf16(q1f[1], ld_fragB(r1 + 32, hf), a1);
      a2 = wmma_bf16(q2f[0], ld_fragB(r2, hf), a2);
      a2 = wmma_bf16(q2f[1], ld_fragB(r2 + 32, hf), a2);
      st1[t] = a1;
      st2[t] = a2;
    }

    softmax_update(st1, m1, l1, o1, &pbuf[wid][0][0], lane16, hf);
    softmax_update(st2, m2, l2, o2, &pbuf[wid][1][0], lane16, hf);

    // in-wave LDS RAW: p writes (D-layout) -> p reads (A-layout)
    asm volatile("s_wait_dscnt 0x0" ::: "memory");

    v16bf p1A = ld_fragA(&pbuf[wid][0][lane16 * 32], hf);
    v16bf p2A = ld_fragA(&pbuf[wid][1][lane16 * 32], hf);
#pragma unroll
    for (int dt = 0; dt < 4; ++dt) {
      v16bf vB = ld_fragB(&vTs[(dt * 16 + lane16) * 32], hf);
      o1[dt] = wmma_bf16(p1A, vB, o1[dt]);
      o2[dt] = wmma_bf16(p2A, vB, o2[dt]);
    }
  }

#pragma unroll
  for (int r = 0; r < 8; ++r) {
    const int row = qrow0 + r + 8 * hf;
    const float inv1 = __frcp_rn(l1[r]);
    const float inv2 = __frcp_rn(l2[r]);
#pragma unroll
    for (int dt = 0; dt < 4; ++dt) {
      const int dd = dt * 16 + lane16;
      const float val = o1[dt][r] * inv1 - lam_h * (o2[dt][r] * inv2);
      outh[(size_t)row * DH + dd] = val;
    }
  }
}

// ---------------------------------------------------------------------------
// GroupNorm over (TQ,DH) per (b,h); writes bf16 in token-major [B*TQ, H*DH].
// ---------------------------------------------------------------------------
__global__ void __launch_bounds__(256)
groupnorm_kernel(const float* __restrict__ att, const float* __restrict__ gamma,
                 const float* __restrict__ beta, __bf16* __restrict__ normed) {
  using namespace cfg;
  __shared__ float red[256];
  const int bh = blockIdx.x, b = bh >> 4, h = bh & 15;
  const float* src = att + (size_t)bh * TQ * DH;
  constexpr int Ntot = TQ * DH;

  float s = 0.f, ss = 0.f;
  for (int i = threadIdx.x; i < Ntot; i += 256) { float x = src[i]; s += x; ss += x * x; }
  red[threadIdx.x] = s; __syncthreads();
  for (int off = 128; off > 0; off >>= 1) {
    if ((int)threadIdx.x < off) red[threadIdx.x] += red[threadIdx.x + off];
    __syncthreads();
  }
  const float mean = red[0] * (1.f / Ntot);
  __syncthreads();
  red[threadIdx.x] = ss; __syncthreads();
  for (int off = 128; off > 0; off >>= 1) {
    if ((int)threadIdx.x < off) red[threadIdx.x] += red[threadIdx.x + off];
    __syncthreads();
  }
  const float var = red[0] * (1.f / Ntot) - mean * mean;
  const float rstd = rsqrtf(var + 1e-5f);
  for (int i = threadIdx.x; i < Ntot; i += 256) {
    const int t = i >> 6, d = i & 63;
    const float y = (src[i] - mean) * rstd * gamma[h * 64 + d] + beta[h * 64 + d];
    normed[((size_t)(b * TQ + t)) * HD + h * 64 + d] = f2bf(y);
  }
}

// ---------------------------------------------------------------------------
extern "C" void kernel_launch(void* const* d_in, const int* in_sizes, int n_in,
                              void* d_out, int out_size, void* d_ws, size_t ws_size,
                              hipStream_t stream) {
  using namespace cfg;
  (void)in_sizes; (void)n_in; (void)out_size; (void)ws_size;

  const float* x    = (const float*)d_in[0];
  const float* enc  = (const float*)d_in[1];
  const float* Q1w  = (const float*)d_in[2];
  const float* Q1b  = (const float*)d_in[3];
  const float* Q2w  = (const float*)d_in[4];
  const float* Q2b  = (const float*)d_in[5];
  const float* K1w  = (const float*)d_in[6];
  const float* K1b  = (const float*)d_in[7];
  const float* K2w  = (const float*)d_in[8];
  const float* K2b  = (const float*)d_in[9];
  const float* Vw   = (const float*)d_in[10];
  const float* Vb   = (const float*)d_in[11];
  const float* lam  = (const float*)d_in[12];
  const float* gng  = (const float*)d_in[13];
  const float* gnb  = (const float*)d_in[14];
  const float* Ow   = (const float*)d_in[15];
  const float* Ob   = (const float*)d_in[16];

  char* wp = (char*)d_ws;
  auto alloc = [&](size_t bytes) -> void* {
    void* r = (void*)wp;
    wp += (bytes + 255) & ~(size_t)255;
    return r;
  };

  const size_t TOK = (size_t)Mtok * E;            // 4 Mi elements
  const size_t WEL = (size_t)HD * E;              // 1 Mi elements
  const size_t PRJ = (size_t)Bsz * H * TQ * DH;   // 4 Mi elements

  __bf16* xb  = (__bf16*)alloc(TOK * 2);
  __bf16* eb  = (__bf16*)alloc(TOK * 2);
  __bf16* w1  = (__bf16*)alloc(WEL * 2);
  __bf16* w2  = (__bf16*)alloc(WEL * 2);
  __bf16* w3  = (__bf16*)alloc(WEL * 2);
  __bf16* w4  = (__bf16*)alloc(WEL * 2);
  __bf16* w5  = (__bf16*)alloc(WEL * 2);
  __bf16* wo  = (__bf16*)alloc(WEL * 2);
  __bf16* q1p = (__bf16*)alloc(PRJ * 2);
  __bf16* q2p = (__bf16*)alloc(PRJ * 2);
  __bf16* k1p = (__bf16*)alloc(PRJ * 2);
  __bf16* k2p = (__bf16*)alloc(PRJ * 2);
  __bf16* vp  = (__bf16*)alloc(PRJ * 2);
  float*  att = (float*)alloc(PRJ * 4);
  __bf16* nrm = (__bf16*)alloc(TOK * 2);

  const int CB = 256;
  cvt_f32_bf16<<<(int)((TOK + CB - 1) / CB), CB, 0, stream>>>(x,   xb, (int)TOK);
  cvt_f32_bf16<<<(int)((TOK + CB - 1) / CB), CB, 0, stream>>>(enc, eb, (int)TOK);
  cvt_f32_bf16<<<(int)((WEL + CB - 1) / CB), CB, 0, stream>>>(Q1w, w1, (int)WEL);
  cvt_f32_bf16<<<(int)((WEL + CB - 1) / CB), CB, 0, stream>>>(Q2w, w2, (int)WEL);
  cvt_f32_bf16<<<(int)((WEL + CB - 1) / CB), CB, 0, stream>>>(K1w, w3, (int)WEL);
  cvt_f32_bf16<<<(int)((WEL + CB - 1) / CB), CB, 0, stream>>>(K2w, w4, (int)WEL);
  cvt_f32_bf16<<<(int)((WEL + CB - 1) / CB), CB, 0, stream>>>(Vw,  w5, (int)WEL);
  cvt_f32_bf16<<<(int)((WEL + CB - 1) / CB), CB, 0, stream>>>(Ow,  wo, (int)WEL);

  // softmax scale folded into q projections, in base-2 domain:
  // scores s' = (q.k)*0.125*log2(e)  ->  softmax = 2^(s'-m) / sum
  const float SCL = 0.125f * 1.44269504088896340736f;

  dim3 gg(HD / 128, Mtok / 128);   // (8, 32)
  gemm_bf16_128x128<0><<<gg, 256, 0, stream>>>(xb, w1, Q1b, q1p, SCL);
  gemm_bf16_128x128<0><<<gg, 256, 0, stream>>>(xb, w2, Q2b, q2p, SCL);
  gemm_bf16_128x128<0><<<gg, 256, 0, stream>>>(eb, w3, K1b, k1p, 1.0f);
  gemm_bf16_128x128<0><<<gg, 256, 0, stream>>>(eb, w4, K2b, k2p, 1.0f);
  gemm_bf16_128x128<0><<<gg, 256, 0, stream>>>(eb, w5, Vb,  vp,  1.0f);

  diff_attn_kernel<<<dim3(Bsz * H, TQ / 128), 256, 0, stream>>>(q1p, q2p, k1p, k2p, vp, lam, att);

  groupnorm_kernel<<<Bsz * H, 256, 0, stream>>>(att, gng, gnb, nrm);

  gemm_bf16_128x128<1><<<dim3(E / 128, Mtok / 128), 256, 0, stream>>>(nrm, wo, Ob, d_out, 1.0f);
}
